// RelPositionMultiHeadedAttention_29686813950157
// MI455X (gfx1250) — compile-verified, matrix operand order checked by reading, not run on hardware
//
#include <hip/hip_runtime.h>
#include <hip/hip_bf16.h>
#include <math.h>

// Problem constants (from reference)
#define BATCH 8
#define T_SEQ 512
#define FDIM  1024
#define NHEAD 16
#define DKH   64
#define PLEN  1023   // 2*T-1

typedef __attribute__((ext_vector_type(16))) _Float16     v16h;
typedef __attribute__((ext_vector_type(8)))  float        v8f;
typedef __attribute__((ext_vector_type(4))) unsigned int  v4u;

// ---------------------------------------------------------------------------
// WMMA helpers (CDNA5 f16 -> f32, 16x16x32)
// ---------------------------------------------------------------------------
__device__ __forceinline__ v8f wmma_f16(v16h a, v16h b, v8f c) {
  // (neg_a, A, neg_b, B, c_mod, C, reuse_a, reuse_b)
  return __builtin_amdgcn_wmma_f32_16x16x32_f16(false, a, false, b, (short)0, c,
                                                false, false);
}

// Load a 16x32 f16 fragment (A layout == B layout for wave32):
// lane l -> row (row0 + (l&15)), 16 consecutive halves at k0 + ((l>>4)*16).
__device__ __forceinline__ v16h frag_ptr(const _Float16* p) {
  union { v16h h; v4u u[2]; } r;
  r.u[0] = ((const v4u*)p)[0];
  r.u[1] = ((const v4u*)p)[1];
  return r.h;
}
__device__ __forceinline__ v16h frag_load(const _Float16* base, int row0,
                                          int ld, int k0) {
  int lane = threadIdx.x & 31;
  return frag_ptr(base + (size_t)(row0 + (lane & 15)) * ld + k0 +
                  ((lane >> 4) << 4));
}

// ---------------------------------------------------------------------------
// Tensor Data Mover: 2-D f16 tile Global -> LDS (CDNA5 TDM, §8 of ISA guide).
// Descriptor built per D# group0/group1 layout; groups 2/3 NULL-equivalent
// (zeros) for 2-D tensors. Reads past tensor_dim return zero (free OOB guard).
// This toolchain exposes the 6-arg builtin (clang-23 / therock headers):
//   (uint32x4 g0, int32x8 g1, int32x4, int32x4, int32x8, i32 cpol)
// ---------------------------------------------------------------------------
#if __has_builtin(__builtin_amdgcn_tensor_load_to_lds)
#define HAVE_TDM 1
typedef __attribute__((ext_vector_type(4))) unsigned int u32x4;
typedef __attribute__((ext_vector_type(8))) int          i32x8;
typedef __attribute__((ext_vector_type(4))) int          i32x4;

__device__ __forceinline__ void tdm_load_2d_f16(
    unsigned lds_off, const void* gaddr, unsigned tensor_d0, unsigned tensor_d1,
    unsigned tile_d0, unsigned tile_d1, unsigned stride0_elems) {
  const unsigned long long ga = (unsigned long long)(uintptr_t)gaddr;
  u32x4 g0;
  g0[0] = 1u;                                            // count=1 (valid D#)
  g0[1] = lds_off;                                       // lds_addr (bytes)
  g0[2] = (unsigned)(ga & 0xFFFFFFFFu);                  // global_addr[31:0]
  g0[3] = (unsigned)((ga >> 32) & 0x1FFFFFFu) | (2u << 30);  // addr[56:32]|type=2
  i32x8 g1;
  g1[0] = (int)(1u << 16);                               // data_size=1 (2B)
  g1[1] = (int)((tensor_d0 & 0xFFFFu) << 16);            // tensor_dim0 lo
  g1[2] = (int)(((tensor_d0 >> 16) & 0xFFFFu) | ((tensor_d1 & 0xFFFFu) << 16));
  g1[3] = (int)(((tensor_d1 >> 16) & 0xFFFFu) | ((tile_d0 & 0xFFFFu) << 16));
  g1[4] = (int)(tile_d1 & 0xFFFFu);                      // tile_dim1, tile_dim2=0
  g1[5] = (int)stride0_elems;                            // tensor_dim0_stride lo
  g1[6] = 0;
  g1[7] = 0;
  i32x4 z4 = {0, 0, 0, 0};
  i32x8 z8 = {0, 0, 0, 0, 0, 0, 0, 0};
  __builtin_amdgcn_tensor_load_to_lds(g0, g1, z4, z4, z8, 0);
}
#else
#define HAVE_TDM 0
#endif

// ---------------------------------------------------------------------------
// Generic projection GEMM:  C[M,FDIM] = A[M,FDIM] @ W[FDIM,FDIM]^T (+ bias)
// A is fp32 (A32) or f16 (A16). Epilogue can write:
//   outA (f16) = acc + bias + pu      (q_u / k / v / p)
//   outB (f16) = acc + bias + pv      (q_v)
//   outC (f32) = acc + bias           (final output)
// Block: 256 threads (8 waves). Tile: 128(M) x 64(N) x 64(K).
// ---------------------------------------------------------------------------
__global__ __launch_bounds__(256) void gemm_wmma(
    const float* __restrict__ A32, const _Float16* __restrict__ A16,
    const float* __restrict__ W, const float* __restrict__ bias,
    const float* __restrict__ pu, const float* __restrict__ pv,
    _Float16* __restrict__ outA, _Float16* __restrict__ outB,
    float* __restrict__ outC, int M) {
  __shared__ _Float16 sA[128][64];   // 16 KB
  __shared__ _Float16 sW[64][64];    //  8 KB

  const int tid  = threadIdx.x;
  const int wave = tid >> 5;
  const int lane = tid & 31;
  const int m0 = blockIdx.x * 128;
  const int n0 = blockIdx.y * 64;

  const int arow = tid >> 1, acs = (tid & 1) * 32;   // 2 threads/row, 32 elems
  const int wrow = tid >> 2, wcs = (tid & 3) * 16;   // 4 threads/row, 16 elems

  v8f acc[4] = {};

  for (int k0 = 0; k0 < FDIM; k0 += 64) {
    // ---- stage A tile (convert fp32 -> f16 if needed) ----
    const bool ok = (m0 + arow) < M;
    if (A16) {
      v4u* d = (v4u*)&sA[arow][acs];
      if (ok) {
        const v4u* g = (const v4u*)(A16 + (size_t)(m0 + arow) * FDIM + k0 + acs);
#pragma unroll
        for (int i = 0; i < 4; ++i) d[i] = g[i];
      } else {
        v4u z = {0u, 0u, 0u, 0u};
#pragma unroll
        for (int i = 0; i < 4; ++i) d[i] = z;
      }
    } else {
      if (ok) {
        const float* g = A32 + (size_t)(m0 + arow) * FDIM + k0 + acs;
#pragma unroll
        for (int i = 0; i < 32; ++i) sA[arow][acs + i] = (_Float16)g[i];
      } else {
#pragma unroll
        for (int i = 0; i < 32; ++i) sA[arow][acs + i] = (_Float16)0.f;
      }
    }
    // ---- stage W tile (rows n0..n0+63 of W, row-major == B fragment) ----
    {
      const float* g = W + (size_t)(n0 + wrow) * FDIM + k0 + wcs;
#pragma unroll
      for (int i = 0; i < 16; ++i) sW[wrow][wcs + i] = (_Float16)g[i];
    }
    // prefetch next A tile into cache hierarchy
    if (k0 + 64 < FDIM && ok) {
      if (A16) __builtin_prefetch(A16 + (size_t)(m0 + arow) * FDIM + k0 + 64 + acs, 0, 1);
      else     __builtin_prefetch(A32 + (size_t)(m0 + arow) * FDIM + k0 + 64 + acs, 0, 1);
    }
    __syncthreads();

    v16h af0 = frag_load(&sA[0][0], wave * 16, 64, 0);
    v16h af1 = frag_load(&sA[0][0], wave * 16, 64, 32);
#pragma unroll
    for (int i = 0; i < 4; ++i) {
      acc[i] = wmma_f16(af0, frag_load(&sW[0][0], i * 16, 64, 0), acc[i]);
      acc[i] = wmma_f16(af1, frag_load(&sW[0][0], i * 16, 64, 32), acc[i]);
    }
    __syncthreads();
  }

  // ---- epilogue ----
#pragma unroll
  for (int i = 0; i < 4; ++i) {
#pragma unroll
    for (int j = 0; j < 8; ++j) {
      const int r = wave * 16 + j + ((lane >> 4) << 3);
      const int m = m0 + r;
      const int n = n0 + i * 16 + (lane & 15);
      if (m < M) {
        float val = acc[i][j] + (bias ? bias[n] : 0.f);
        if (outA) outA[(size_t)m * FDIM + n] = (_Float16)(val + (pu ? pu[n] : 0.f));
        if (outB) outB[(size_t)m * FDIM + n] = (_Float16)(val + (pv ? pv[n] : 0.f));
        if (outC) outC[(size_t)m * FDIM + n] = val;
      }
    }
  }
}

// ---------------------------------------------------------------------------
// Fused rel-pos flash attention.
//   scores[t,s] = ( qu[t]·k[s] + qv[t]·p[s - t + 511] ) / 8,   softmax over s,
//   x[t,:] = attn @ v.
// One block = (b, h, 64 t-rows). 128 threads = 4 waves, each wave owns 16 rows.
// The bd term only needs a 31-wide diagonal band per 16x16 tile: compute a
// 16x32 WMMA tile against p-rows [pstart, pstart+32) and gather bd[r, c-r+15].
// K-chunk and p-window tiles are staged via the Tensor Data Mover when the
// builtin is available (TDM OOB-zeroing also implements the p-row guard).
// ---------------------------------------------------------------------------
__global__ __launch_bounds__(128) void attn_kernel(
    const _Float16* __restrict__ qu, const _Float16* __restrict__ qv,
    const _Float16* __restrict__ kf, const _Float16* __restrict__ vf,
    const _Float16* __restrict__ pf, const unsigned char* __restrict__ mask,
    _Float16* __restrict__ xf) {
  __shared__ _Float16 sK[64][DKH];    // K chunk, row-major (s, d)     8 KB
  __shared__ _Float16 sVt[DKH][64];   // V chunk transposed (d, s)     8 KB
  __shared__ _Float16 sP[128][DKH];   // p window rows                16 KB
  __shared__ float    sBD[4][16][32]; // per-wave bd band / pexp alias 8 KB
  __shared__ float    sS[4][16][64];  // per-wave score strip         16 KB
  __shared__ float    sMst[64], sLst[64], sAl[64];

  const int tid  = threadIdx.x;
  const int wave = tid >> 5;
  const int lane = tid & 31;
  const int blk = blockIdx.x;
  const int tb = blk & 7;           // T/64
  const int h  = (blk >> 3) & 15;
  const int b  = blk >> 7;
  const int t0 = tb * 64;

  if (tid < 64) { sMst[tid] = -INFINITY; sLst[tid] = 0.f; }

  // qu / qv fragments for this wave's 16 t-rows (K = DK = 64 -> two chunks)
  const int r0 = t0 + wave * 16;
  const _Float16* qub = qu + (size_t)(b * T_SEQ + r0) * FDIM + h * DKH;
  const _Float16* qvb = qv + (size_t)(b * T_SEQ + r0) * FDIM + h * DKH;
  v16h qu0 = frag_load(qub, 0, FDIM, 0);
  v16h qu1 = frag_load(qub, 0, FDIM, 32);
  v16h qv0 = frag_load(qvb, 0, FDIM, 0);
  v16h qv1 = frag_load(qvb, 0, FDIM, 32);

  v8f o[4] = {};  // 16 t-rows x 64 d-cols accumulator

  for (int s0 = 0; s0 < T_SEQ; s0 += 64) {
    __syncthreads();  // prior iteration done consuming LDS tiles

    const int jbase = s0 - t0 + 448;  // = s0 - t0 - 63 + 511, always >= 0

#if HAVE_TDM
    // ---- TDM: async DMA of K chunk and p window into LDS (wave 0 issues) ----
    if (wave == 0) {
      tdm_load_2d_f16((unsigned)(uintptr_t)&sK[0][0],
                      kf + (size_t)(b * T_SEQ + s0) * FDIM + h * DKH,
                      DKH, 64, DKH, 64, FDIM);
      tdm_load_2d_f16((unsigned)(uintptr_t)&sP[0][0],
                      pf + (size_t)jbase * FDIM + h * DKH,
                      DKH, (unsigned)(PLEN - jbase),  // OOB rows -> zero
                      DKH, 128, FDIM);
    }
#else
    {  // K chunk: 2 threads per row, 32 halves each
      const int row = tid >> 1, cb = (tid & 1) * 32;
      const v4u* g = (const v4u*)(kf + (size_t)(b * T_SEQ + s0 + row) * FDIM +
                                  h * DKH + cb);
      v4u* d = (v4u*)&sK[row][cb];
      d[0] = g[0]; d[1] = g[1]; d[2] = g[2]; d[3] = g[3];
    }
    {  // p window: one row per thread, zero-fill OOB rows
      const int j = jbase + tid;
      v4u* d = (v4u*)&sP[tid][0];
      if (j < PLEN) {
        const v4u* g = (const v4u*)(pf + (size_t)j * FDIM + h * DKH);
#pragma unroll
        for (int i = 0; i < 8; ++i) d[i] = g[i];
      } else {
        v4u z = {0u, 0u, 0u, 0u};
#pragma unroll
        for (int i = 0; i < 8; ++i) d[i] = z;
      }
    }
#endif

    {  // V chunk, transposed into sVt[d][s] (TDM cannot transpose)
      const int row = tid >> 1, cb = (tid & 1) * 32;
      const _Float16* g = vf + (size_t)(b * T_SEQ + s0 + row) * FDIM + h * DKH + cb;
#pragma unroll
      for (int i = 0; i < 32; ++i) sVt[cb + i][row] = g[i];
    }

#if HAVE_TDM
    if (wave == 0) __builtin_amdgcn_s_wait_tensorcnt(0);
#endif
    __syncthreads();

    // ---- scores: ac (WMMA) + banded bd (WMMA) ----
#pragma unroll
    for (int si = 0; si < 4; ++si) {
      v8f acw = {};
      acw = wmma_f16(qu0, frag_load(&sK[0][0], si * 16, DKH, 0), acw);
      acw = wmma_f16(qu1, frag_load(&sK[0][0], si * 16, DKH, 32), acw);

      const int pstart = si * 16 - wave * 16 + 48;  // in [0, 96]
      v8f bd0 = {}, bd1 = {};
      bd0 = wmma_f16(qv0, frag_load(&sP[0][0], pstart, DKH, 0), bd0);
      bd0 = wmma_f16(qv1, frag_load(&sP[0][0], pstart, DKH, 32), bd0);
      bd1 = wmma_f16(qv0, frag_load(&sP[0][0], pstart + 16, DKH, 0), bd1);
      bd1 = wmma_f16(qv1, frag_load(&sP[0][0], pstart + 16, DKH, 32), bd1);

#pragma unroll
      for (int j = 0; j < 8; ++j) {  // stage bd band in per-wave LDS
        const int r = j + ((lane >> 4) << 3);
        sBD[wave][r][lane & 15] = bd0[j];
        sBD[wave][r][(lane & 15) + 16] = bd1[j];
      }
#pragma unroll
      for (int j = 0; j < 8; ++j) {  // gather diagonal + combine + mask
        const int r = j + ((lane >> 4) << 3);
        const int c = lane & 15;
        float sc = (acw[j] + sBD[wave][r][c - r + 15]) * 0.125f;
        const int scol = si * 16 + c;
        if (!mask[b * T_SEQ + s0 + scol]) sc = -3.0e38f;
        sS[wave][r][scol] = sc;
      }
    }

    // ---- online softmax over this 64-col strip (lane r < 16 owns row r) ----
    if (lane < 16) {
      const int r = lane, row = wave * 16 + r;
      float mo = sMst[row], lo = sLst[row];
      float mx = mo;
#pragma unroll 8
      for (int c = 0; c < 64; ++c) mx = fmaxf(mx, sS[wave][r][c]);
      const float alpha = __expf(mo - mx);
      _Float16* pexp = (_Float16*)&sBD[wave][0][0];  // 16x64 f16 alias
      float ssum = 0.f;
#pragma unroll 8
      for (int c = 0; c < 64; ++c) {
        const float e = __expf(sS[wave][r][c] - mx);
        ssum += e;
        pexp[r * 64 + c] = (_Float16)e;
      }
      sMst[row] = mx;
      sLst[row] = lo * alpha + ssum;
      sAl[row]  = alpha;
    }

    // ---- rescale accumulator + P@V (WMMA) ----
    float a8[8];
#pragma unroll
    for (int j = 0; j < 8; ++j) a8[j] = sAl[wave * 16 + j + ((lane >> 4) << 3)];
    const _Float16* pexp = (const _Float16*)&sBD[wave][0][0];
    v16h pa0 = frag_load(pexp, 0, 64, 0);
    v16h pa1 = frag_load(pexp, 0, 64, 32);
#pragma unroll
    for (int i = 0; i < 4; ++i) {
      v8f t = o[i];
#pragma unroll
      for (int j = 0; j < 8; ++j) t[j] *= a8[j];
      t = wmma_f16(pa0, frag_load(&sVt[0][0], i * 16, 64, 0), t);
      t = wmma_f16(pa1, frag_load(&sVt[0][0], i * 16, 64, 32), t);
      o[i] = t;
    }
  }

  // ---- finalize: x = o / l, write f16 (B*T, F) ----
#pragma unroll
  for (int i = 0; i < 4; ++i) {
#pragma unroll
    for (int j = 0; j < 8; ++j) {
      const int r = j + ((lane >> 4) << 3);
      const int trow = t0 + wave * 16 + r;
      const float inv = 1.0f / sLst[wave * 16 + r];
      xf[(size_t)(b * T_SEQ + trow) * FDIM + h * DKH + i * 16 + (lane & 15)] =
          (_Float16)(o[i][j] * inv);
    }
  }
}

// ---------------------------------------------------------------------------
// Launch: QKV + pos projections -> fused attention -> output projection.
// Workspace layout (f16): qu, qv, k, v (8 MB each), p (~2 MB), x (8 MB) = 42 MB.
// ---------------------------------------------------------------------------
extern "C" void kernel_launch(void* const* d_in, const int* in_sizes, int n_in,
                              void* d_out, int out_size, void* d_ws,
                              size_t ws_size, hipStream_t stream) {
  const float* query   = (const float*)d_in[0];
  const float* key     = (const float*)d_in[1];
  const float* value   = (const float*)d_in[2];
  const unsigned char* mask = (const unsigned char*)d_in[3];
  const float* pos_emb = (const float*)d_in[4];
  // d_in[5], d_in[6]: empty kv caches (0 elements) -> ignored
  const float* Wq   = (const float*)d_in[7];
  const float* bq   = (const float*)d_in[8];
  const float* Wk   = (const float*)d_in[9];
  const float* bk   = (const float*)d_in[10];
  const float* Wv   = (const float*)d_in[11];
  const float* bv   = (const float*)d_in[12];
  const float* Wpos = (const float*)d_in[13];
  const float* pbu  = (const float*)d_in[14];
  const float* pbv  = (const float*)d_in[15];
  const float* Wout = (const float*)d_in[16];
  const float* bout = (const float*)d_in[17];
  float* out = (float*)d_out;

  char* ws = (char*)d_ws;
  const size_t SZ_BTF = (size_t)BATCH * T_SEQ * FDIM * sizeof(_Float16);  // 8 MB
  _Float16* quf = (_Float16*)ws; ws += SZ_BTF;
  _Float16* qvf = (_Float16*)ws; ws += SZ_BTF;
  _Float16* kff = (_Float16*)ws; ws += SZ_BTF;
  _Float16* vff = (_Float16*)ws; ws += SZ_BTF;
  _Float16* pff = (_Float16*)ws; ws += (size_t)PLEN * FDIM * sizeof(_Float16);
  _Float16* xff = (_Float16*)ws; ws += SZ_BTF;

  const int M = BATCH * T_SEQ;  // 4096
  dim3 gBig((M + 127) / 128, FDIM / 64);       // 32 x 16
  dim3 gPos((PLEN + 127) / 128, FDIM / 64);    // 8 x 16

  // Q projection: writes q_u and q_v in one pass (bias + pos_bias fused)
  gemm_wmma<<<gBig, 256, 0, stream>>>(query, nullptr, Wq, bq, pbu, pbv,
                                      quf, qvf, nullptr, M);
  gemm_wmma<<<gBig, 256, 0, stream>>>(key, nullptr, Wk, bk, nullptr, nullptr,
                                      kff, nullptr, nullptr, M);
  gemm_wmma<<<gBig, 256, 0, stream>>>(value, nullptr, Wv, bv, nullptr, nullptr,
                                      vff, nullptr, nullptr, M);
  gemm_wmma<<<gPos, 256, 0, stream>>>(pos_emb, nullptr, Wpos, nullptr, nullptr,
                                      nullptr, pff, nullptr, nullptr, PLEN);

  // Fused rel-pos flash attention: B*H*(T/64) = 1024 blocks of 128 threads
  attn_kernel<<<dim3(BATCH * NHEAD * (T_SEQ / 64)), 128, 0, stream>>>(
      quf, qvf, kff, vff, pff, mask, xff);

  // Output projection (f16 activations, fp32 result)
  gemm_wmma<<<gBig, 256, 0, stream>>>(nullptr, xff, Wout, bout, nullptr,
                                      nullptr, nullptr, nullptr, out, M);
}
